// KwisatzHaderach_24017457119651
// MI455X (gfx1250) — matrix operand order repaired and checked by reading, not compiled, not run on hardware
//
#include <hip/hip_runtime.h>
#include <hip/hip_bf16.h>
#include <stdint.h>

// Problem constants (match reference)
#define NPART 20000
#define KNBR  48
#define INV_RADIUS (1.0f / 0.225f)
#define NTB 4   // n-tiles per wave in the WMMA GEMM (A-fragment reuse)

// WMMA vector types (CDNA5, wave32)
typedef __attribute__((ext_vector_type(16))) __bf16 v16bf;
typedef __attribute__((ext_vector_type(8)))  float  v8f;

__device__ __forceinline__ uint16_t f2bf(float f) {
  uint32_t u = __float_as_uint(f);
  u += 0x7FFFu + ((u >> 16) & 1u);          // round-to-nearest-even
  return (uint16_t)(u >> 16);
}
__device__ __forceinline__ float bf2f(uint16_t v) {
  return __uint_as_float(((uint32_t)v) << 16);
}
__device__ __forceinline__ float sgnf(float x) {
  return (x > 0.f) ? 1.f : ((x < 0.f) ? -1.f : 0.f);
}

// ---------------------------------------------------------------- feats = [vel, mass]
__global__ void build_feats_kernel(const float* __restrict__ vel,
                                   const float* __restrict__ mass,
                                   float* __restrict__ feats) {
  int i = blockIdx.x * blockDim.x + threadIdx.x;
  if (i >= NPART) return;
  feats[4 * i + 0] = vel[3 * i + 0];
  feats[4 * i + 1] = vel[3 * i + 1];
  feats[4 * i + 2] = vel[3 * i + 2];
  feats[4 * i + 3] = mass[i];
}

// --------------------------------------------- per-pair geometry: 8 cells + 8 weights
__global__ void prep_pairs_kernel(const float* __restrict__ pos,
                                  const int* __restrict__ nbr_idx,
                                  const unsigned char* __restrict__ nbr_mask,
                                  unsigned long long* __restrict__ cells_out,
                                  float* __restrict__ w_out) {
  int p = blockIdx.x * blockDim.x + threadIdx.x;
  if (p >= NPART * KNBR) return;
  int i = p / KNBR;
  int j = nbr_idx[p];
  float x = (pos[3 * j + 0] - pos[3 * i + 0]) * INV_RADIUS;
  float y = (pos[3 * j + 1] - pos[3 * i + 1]) * INV_RADIUS;
  float z = (pos[3 * j + 2] - pos[3 * i + 2]) * INV_RADIUS;
  float r2 = x * x + y * y + z * z;
  float t = 1.0f - r2;
  float win = fminf(fmaxf(t * t * t, 0.0f), 1.0f);
  if (!nbr_mask[p]) win = 0.0f;

  // ball -> cube (volume preserving), as in reference
  float norm = sqrtf(r2);
  float xy2  = x * x + y * y;
  bool  polar = (1.25f * z * z > xy2);
  float s_pol = sqrtf(3.0f * norm / (norm + fabsf(z) + 1e-20f));
  float s_eq  = norm * rsqrtf(xy2 + 1e-20f);
  float s  = polar ? s_pol : s_eq;
  float zc = polar ? (sgnf(z) * norm) : (1.5f * z);
  bool  nz = r2 > 0.0f;
  float xc = nz ? x * s : 0.0f;
  float yc = nz ? y * s : 0.0f;
  zc = nz ? zc : 0.0f;
  float nxy = sqrtf(xc * xc + yc * yc);
  bool  cond = fabsf(yc) <= fabsf(xc);
  float sx = sgnf(xc) * nxy;
  float sy = sgnf(yc) * nxy;
  float atan_yx = atanf(yc / (fabsf(xc) > 0.0f ? xc : 1.0f));
  float atan_xy = atanf(xc / (fabsf(yc) > 0.0f ? yc : 1.0f));
  const float fop = 4.0f / 3.14159265358979323846f;
  float u = cond ? sx : sy * fop * atan_xy;
  float v = cond ? sx * fop * atan_yx : sy;
  if (nxy <= 0.0f) { u = xc; v = yc; }

  // trilinear cell coordinates on 4^3 grid, align_corners
  float cu = fminf(fmaxf((u  + 1.0f) * 1.5f, 0.0f), 3.0f);
  float cv = fminf(fmaxf((v  + 1.0f) * 1.5f, 0.0f), 3.0f);
  float cw = fminf(fmaxf((zc + 1.0f) * 1.5f, 0.0f), 3.0f);
  int ix0 = (int)floorf(cu), iy0 = (int)floorf(cv), iz0 = (int)floorf(cw);
  float fx = cu - (float)ix0, fy = cv - (float)iy0, fz = cw - (float)iz0;
  int ix1 = min(ix0 + 1, 3), iy1 = min(iy0 + 1, 3), iz1 = min(iz0 + 1, 3);

  unsigned long long cellpack = 0;
  float wv[8];
  int c = 0;
  for (int dx = 0; dx < 2; ++dx) {
    float wx = dx ? fx : 1.0f - fx; int ix = dx ? ix1 : ix0;
    for (int dy = 0; dy < 2; ++dy) {
      float wy = dy ? fy : 1.0f - fy; int iy = dy ? iy1 : iy0;
      for (int dz = 0; dz < 2; ++dz) {
        float wz = dz ? fz : 1.0f - fz; int iz = dz ? iz1 : iz0;
        int cell = (ix * 4 + iy) * 4 + iz;
        cellpack |= ((unsigned long long)cell) << (8 * c);
        wv[c] = win * wx * wy * wz;
        ++c;
      }
    }
  }
  cells_out[p] = cellpack;
#pragma unroll
  for (int q = 0; q < 8; ++q) w_out[(size_t)p * 8 + q] = wv[q];
}

// ----------------- pack B = [ conv W (g,c,o) as cols g*Cout+o | dense W cols ] -> bf16
// Packed in contiguous-k B-fragment order (16-bit B 32x16), matching pack_A:
// lane 0-15 : N = lane, K = 0..15 (2 per VGPR);  lane 16-31 : K = 16..31.
__global__ void pack_B_kernel(const float* __restrict__ convW,
                              const float* __restrict__ denseW,
                              uint32_t* __restrict__ out,
                              int Cin, int Cout, int kblocks, int ntiles, int Ncols) {
  int t = blockIdx.x * blockDim.x + threadIdx.x;
  int total = kblocks * ntiles * 256;
  if (t >= total) return;
  int v    = t & 7;
  int lane = (t >> 3) & 31;
  int tile = t >> 8;
  int nt = tile % ntiles;
  int kb = tile / ntiles;
  int n  = nt * 16 + (lane & 15);
  int k0 = kb * 32 + ((lane >> 4) & 1) * 16 + 2 * v;
  float lo = 0.f, hi = 0.f;
  if (n < Ncols) {
    int convCols = 64 * Cout;
    if (n < convCols) {
      int g = n / Cout, o = n % Cout;
      if (k0     < Cin) lo = convW[((size_t)g * Cin + k0    ) * Cout + o];
      if (k0 + 1 < Cin) hi = convW[((size_t)g * Cin + k0 + 1) * Cout + o];
    } else {
      int o = n - convCols;
      if (k0     < Cin) lo = denseW[(size_t)(k0    ) * Cout + o];
      if (k0 + 1 < Cin) hi = denseW[(size_t)(k0 + 1) * Cout + o];
    }
  }
  out[t] = ((uint32_t)f2bf(hi) << 16) | (uint32_t)f2bf(lo);
}

// --------- pack A = x (optional ReLU) -> bf16 in contiguous-k A-fragment order (16-bit A 16x32):
// lane 0-15: M=lane, K = 0..15 (2 per VGPR); lane 16-31: K = 16..31.
__global__ void pack_A_kernel(const float* __restrict__ x,
                              uint32_t* __restrict__ out,
                              int Cin, int kblocks, int mtiles, int do_relu) {
  int t = blockIdx.x * blockDim.x + threadIdx.x;
  int total = mtiles * kblocks * 256;
  if (t >= total) return;
  int v    = t & 7;
  int lane = (t >> 3) & 31;
  int tile = t >> 8;
  int kb = tile % kblocks;
  int mt = tile / kblocks;
  int m  = mt * 16 + (lane & 15);
  int klocal = (v < 4) ? (2 * v) : (16 + 2 * (v - 4));
  int k0 = kb * 32 + ((lane >> 4) & 1) * 16 + 2 * v;
  float lo = 0.f, hi = 0.f;
  if (m < NPART) {
    if (k0     < Cin) lo = x[(size_t)m * Cin + k0];
    if (k0 + 1 < Cin) hi = x[(size_t)m * Cin + k0 + 1];
  }
  if (do_relu) { lo = fmaxf(lo, 0.f); hi = fmaxf(hi, 0.f); }
  out[t] = ((uint32_t)f2bf(hi) << 16) | (uint32_t)f2bf(lo);
}

// ------------------------------------------------ WMMA GEMM: Y(bf16) = A(bf16) x B(bf16)
// Each wave owns one 16-row M tile and NTB consecutive 16-col N tiles, reusing the A
// fragments (kept in VGPRs) across all of them.
__global__ void gemm_bf16_kernel(const uint32_t* __restrict__ A,
                                 const uint32_t* __restrict__ B,
                                 uint16_t* __restrict__ Y,
                                 int kblocks, int ntiles, int ldY, int totalWaves) {
  int w = blockIdx.x * (blockDim.x >> 5) + (threadIdx.x >> 5);
  if (w >= totalWaves) return;                    // wave-uniform: EXEC stays all-ones
  int lane = threadIdx.x & 31;
  int nblk = (ntiles + NTB - 1) / NTB;
  int mt  = w / nblk;
  int nt0 = (w % nblk) * NTB;

  const v16bf a0 = *(const v16bf*)(A + ((size_t)(mt * kblocks + 0) * 32 + lane) * 8);
  v16bf a1 = a0;
  if (kblocks > 1)
    a1 = *(const v16bf*)(A + ((size_t)(mt * kblocks + 1) * 32 + lane) * 8);

  int nend = min(nt0 + NTB, ntiles);
  int row0 = mt * 16 + ((lane >> 4) & 1) * 8;
  for (int nt = nt0; nt < nend; ++nt) {           // wave-uniform bound
    v8f c = {};
    {
      const v16bf b = *(const v16bf*)(B + ((size_t)(0 * ntiles + nt) * 32 + lane) * 8);
      c = __builtin_amdgcn_wmma_f32_16x16x32_bf16(false, a0, false, b, (short)0, c,
                                                  false, false);
    }
    if (kblocks > 1) {
      const v16bf b = *(const v16bf*)(B + ((size_t)(1 * ntiles + nt) * 32 + lane) * 8);
      c = __builtin_amdgcn_wmma_f32_16x16x32_bf16(false, a1, false, b, (short)0, c,
                                                  false, false);
    }
    // C/D layout: lanes 0-15 -> M=r, N=lane; lanes 16-31 -> M=8+r, N=lane-16
    int col = nt * 16 + (lane & 15);
#pragma unroll
    for (int r = 0; r < 8; ++r)
      Y[(size_t)(row0 + r) * ldY + col] = f2bf(c[r]);
  }
}

// ------------------------------------------ layer-0 combine: h = [conv0+cb | dense0+db]
__global__ void combine0_kernel(const uint16_t* __restrict__ Y,
                                const unsigned long long* __restrict__ cells,
                                const float* __restrict__ wts,
                                const int* __restrict__ nbr_idx,
                                const float* __restrict__ cb, const float* __restrict__ db,
                                float* __restrict__ h) {
  int wave = (blockIdx.x * blockDim.x + threadIdx.x) >> 5;
  int lane = threadIdx.x & 31;
  if (wave >= NPART) return;
  int i = wave;
  float acc = 0.f;
  for (int kk = 0; kk < KNBR; ++kk) {
    int p = i * KNBR + kk;
    int j = nbr_idx[p];
    unsigned long long cp = cells[p];
    const float* wp = wts + (size_t)p * 8;
    const uint16_t* yb = Y + (size_t)j * 2080;
#pragma unroll
    for (int c = 0; c < 8; ++c) {
      int cell = (int)((cp >> (8 * c)) & 63ull);
      acc += wp[c] * bf2f(yb[cell * 32 + lane]);
    }
  }
  float dv = bf2f(Y[(size_t)i * 2080 + 2048 + lane]);
  h[(size_t)i * 64 + lane]      = acc + cb[lane];
  h[(size_t)i * 64 + 32 + lane] = dv + db[lane];
}

// ----------------------------------- layers 1,2 combine: h_new = conv + dense + h_old
// Each lane owns outputs {2*lane, 2*lane+1}: one aligned u32 load = packed bf16 pair,
// halving VMEM instruction count in the bandwidth-bound inner loop.
__global__ void combine12_kernel(const uint16_t* __restrict__ Y,
                                 const unsigned long long* __restrict__ cells,
                                 const float* __restrict__ wts,
                                 const int* __restrict__ nbr_idx,
                                 const float* __restrict__ cb, const float* __restrict__ db,
                                 const float* __restrict__ h_old, float* __restrict__ h_new) {
  int wave = (blockIdx.x * blockDim.x + threadIdx.x) >> 5;
  int lane = threadIdx.x & 31;
  if (wave >= NPART) return;
  int i = wave;
  int o0 = 2 * lane;
  float acc0 = cb[o0];
  float acc1 = cb[o0 + 1];
  for (int kk = 0; kk < KNBR; ++kk) {
    int p = i * KNBR + kk;
    int j = nbr_idx[p];
    unsigned long long cp = cells[p];
    const float* wp = wts + (size_t)p * 8;
    const uint32_t* yb = (const uint32_t*)(Y + (size_t)j * 4160);
    __builtin_prefetch(wts + (size_t)(p + 1) * 8, 0, 0);
#pragma unroll
    for (int c = 0; c < 8; ++c) {
      int cell = (int)((cp >> (8 * c)) & 63ull);
      float w = wp[c];
      uint32_t pk = yb[cell * 32 + lane];         // coalesced 128B per cell row
      acc0 += w * bf2f((uint16_t)pk);
      acc1 += w * bf2f((uint16_t)(pk >> 16));
    }
  }
  uint32_t dpk = ((const uint32_t*)(Y + (size_t)i * 4160 + 4096))[lane];
  float d0 = bf2f((uint16_t)dpk)         + db[o0];
  float d1 = bf2f((uint16_t)(dpk >> 16)) + db[o0 + 1];
  float r0 = acc0 + d0 + h_old[(size_t)i * 64 + o0];
  float r1 = acc1 + d1 + h_old[(size_t)i * 64 + o0 + 1];
  float2* hv = (float2*)(h_new + (size_t)i * 64);
  hv[lane] = make_float2(r0, r1);
}

// ------------------- layer-3 combine + finalize: disp -> (pos+disp, disp/TIME_STEP)
__global__ void combine3_kernel(const uint16_t* __restrict__ Y,
                                const unsigned long long* __restrict__ cells,
                                const float* __restrict__ wts,
                                const int* __restrict__ nbr_idx,
                                const float* __restrict__ cb, const float* __restrict__ db,
                                const float* __restrict__ pos, float* __restrict__ out) {
  int wave = (blockIdx.x * blockDim.x + threadIdx.x) >> 5;
  int lane = threadIdx.x & 31;
  if (wave >= NPART) return;
  int i = wave;
  float a0 = 0.f, a1 = 0.f, a2 = 0.f;
  for (int t = lane; t < KNBR * 8; t += 32) {
    int kk = t >> 3, c = t & 7;
    int p = i * KNBR + kk;
    int j = nbr_idx[p];
    int cell = (int)((cells[p] >> (8 * c)) & 63ull);
    float w = wts[(size_t)p * 8 + c];
    const uint16_t* yr = Y + (size_t)j * 208 + cell * 3;
    a0 += w * bf2f(yr[0]);
    a1 += w * bf2f(yr[1]);
    a2 += w * bf2f(yr[2]);
  }
  for (int off = 16; off > 0; off >>= 1) {
    a0 += __shfl_down(a0, off, 32);
    a1 += __shfl_down(a1, off, 32);
    a2 += __shfl_down(a2, off, 32);
  }
  if (lane == 0) {
    float a[3] = {a0, a1, a2};
#pragma unroll
    for (int o = 0; o < 3; ++o) {
      float conv = a[o] + cb[o];
      float den  = bf2f(Y[(size_t)i * 208 + 192 + o]) + db[o];
      float disp = conv + den;
      out[(size_t)i * 3 + o] = pos[(size_t)i * 3 + o] + disp;
      out[(size_t)NPART * 3 + (size_t)i * 3 + o] = disp * 100.0f;  // / TIME_STEP
    }
  }
}

// =====================================================================================
extern "C" void kernel_launch(void* const* d_in, const int* in_sizes, int n_in,
                              void* d_out, int out_size, void* d_ws, size_t ws_size,
                              hipStream_t stream) {
  (void)in_sizes; (void)n_in; (void)out_size; (void)ws_size;
  const float* pos  = (const float*)d_in[0];
  const float* vel  = (const float*)d_in[1];
  const float* mass = (const float*)d_in[2];
  const float* c0w = (const float*)d_in[3];  const float* c0b = (const float*)d_in[4];
  const float* d0w = (const float*)d_in[5];  const float* d0b = (const float*)d_in[6];
  const float* c1w = (const float*)d_in[7];  const float* c1b = (const float*)d_in[8];
  const float* d1w = (const float*)d_in[9];  const float* d1b = (const float*)d_in[10];
  const float* c2w = (const float*)d_in[11]; const float* c2b = (const float*)d_in[12];
  const float* d2w = (const float*)d_in[13]; const float* d2b = (const float*)d_in[14];
  const float* c3w = (const float*)d_in[15]; const float* c3b = (const float*)d_in[16];
  const float* d3w = (const float*)d_in[17]; const float* d3b = (const float*)d_in[18];
  const int* nbr_idx = (const int*)d_in[19];
  const unsigned char* nbr_mask = (const unsigned char*)d_in[20];
  float* out = (float*)d_out;

  // ---- carve workspace (all offsets 256B-aligned; v16bf loads need 32B) ----
  char* ws = (char*)d_ws;
  size_t off = 0;
  auto carve = [&](size_t bytes) -> void* {
    void* p = ws + off;
    off += (bytes + 255) & ~(size_t)255;
    return p;
  };
  float*    feats = (float*)carve((size_t)NPART * 4 * 4);
  float*    hA    = (float*)carve((size_t)NPART * 64 * 4);
  float*    hB    = (float*)carve((size_t)NPART * 64 * 4);
  uint32_t* Apack = (uint32_t*)carve((size_t)1250 * 2 * 256 * 4);
  uint32_t* B0    = (uint32_t*)carve((size_t)1 * 130 * 256 * 4);
  uint32_t* B1    = (uint32_t*)carve((size_t)2 * 260 * 256 * 4);
  uint32_t* B2    = (uint32_t*)carve((size_t)2 * 260 * 256 * 4);
  uint32_t* B3    = (uint32_t*)carve((size_t)2 * 13 * 256 * 4);
  unsigned long long* cells = (unsigned long long*)carve((size_t)NPART * KNBR * 8);
  float*    wts   = (float*)carve((size_t)NPART * KNBR * 8 * 4);
  uint16_t* Y     = (uint16_t*)carve((size_t)NPART * 4160 * 2);  // bf16, fits in 192MB L2

  const int mtiles = NPART / 16;  // 1250 (exact)
  const int nblk0 = (130 + NTB - 1) / NTB;  // 33
  const int nblk1 = (260 + NTB - 1) / NTB;  // 65
  const int nblk3 = (13 + NTB - 1) / NTB;   // 4

  // ---- precompute (geometry identical across all 4 layers) ----
  build_feats_kernel<<<(NPART + 255) / 256, 256, 0, stream>>>(vel, mass, feats);
  prep_pairs_kernel<<<(NPART * KNBR + 255) / 256, 256, 0, stream>>>(
      pos, nbr_idx, nbr_mask, cells, wts);
  pack_B_kernel<<<(1 * 130 * 256) / 256, 256, 0, stream>>>(c0w, d0w, B0, 4, 32, 1, 130, 2080);
  pack_B_kernel<<<(2 * 260 * 256) / 256, 256, 0, stream>>>(c1w, d1w, B1, 64, 64, 2, 260, 4160);
  pack_B_kernel<<<(2 * 260 * 256) / 256, 256, 0, stream>>>(c2w, d2w, B2, 64, 64, 2, 260, 4160);
  pack_B_kernel<<<(2 * 13 * 256) / 256, 256, 0, stream>>>(c3w, d3w, B3, 64, 3, 2, 13, 195);

  // ---- layer 0: Cin=4 (zero-padded K), cols = 64*32 conv + 32 dense = 2080 ----
  pack_A_kernel<<<(mtiles * 1 * 256) / 256, 256, 0, stream>>>(feats, Apack, 4, 1, mtiles, 0);
  { int tw = mtiles * nblk0;
    gemm_bf16_kernel<<<(tw + 7) / 8, 256, 0, stream>>>(Apack, B0, Y, 1, 130, 2080, tw); }
  combine0_kernel<<<(NPART * 32) / 256, 256, 0, stream>>>(Y, cells, wts, nbr_idx, c0b, d0b, hA);

  // ---- layer 1: ReLU fused into A-pack; cols = 64*64 + 64 = 4160 ----
  pack_A_kernel<<<(mtiles * 2 * 256) / 256, 256, 0, stream>>>(hA, Apack, 64, 2, mtiles, 1);
  { int tw = mtiles * nblk1;
    gemm_bf16_kernel<<<(tw + 7) / 8, 256, 0, stream>>>(Apack, B1, Y, 2, 260, 4160, tw); }
  combine12_kernel<<<(NPART * 32) / 256, 256, 0, stream>>>(
      Y, cells, wts, nbr_idx, c1b, d1b, hA, hB);

  // ---- layer 2 ----
  pack_A_kernel<<<(mtiles * 2 * 256) / 256, 256, 0, stream>>>(hB, Apack, 64, 2, mtiles, 1);
  { int tw = mtiles * nblk1;
    gemm_bf16_kernel<<<(tw + 7) / 8, 256, 0, stream>>>(Apack, B2, Y, 2, 260, 4160, tw); }
  combine12_kernel<<<(NPART * 32) / 256, 256, 0, stream>>>(
      Y, cells, wts, nbr_idx, c2b, d2b, hB, hA);

  // ---- layer 3: Cout=3, cols = 192 + 3 -> padded to 208 ----
  pack_A_kernel<<<(mtiles * 2 * 256) / 256, 256, 0, stream>>>(hA, Apack, 64, 2, mtiles, 1);
  { int tw = mtiles * nblk3;
    gemm_bf16_kernel<<<(tw + 7) / 8, 256, 0, stream>>>(Apack, B3, Y, 2, 13, 208, tw); }
  combine3_kernel<<<(NPART * 32) / 256, 256, 0, stream>>>(
      Y, cells, wts, nbr_idx, c3b, d3b, pos, out);
}
